// FastrayTransformer_81355270520978
// MI455X (gfx1250) — matrix operand order repaired and checked by reading, not compiled, not run on hardware
//
#include <hip/hip_runtime.h>
#include <math.h>

typedef _Float16     v16h __attribute__((ext_vector_type(16)));
typedef float        v8f  __attribute__((ext_vector_type(8)));
typedef unsigned int v4u  __attribute__((ext_vector_type(4)));
typedef int          v4i  __attribute__((ext_vector_type(4)));
typedef int          v8i  __attribute__((ext_vector_type(8)));

#define B_     2
#define N_     6
#define BN_    12
#define CIN_   256
#define H_     16
#define W_     44
#define HW_    704
#define DD_    59
#define COUT_  64
#define OC_    123      /* DD_ + COUT_ */
#define X_     128
#define Y_     128
#define Z_     7
#define PTILES (HW_ / 16)          /* 44 pixel tiles per view */
#define BRSTRIDE 17                /* Braw row stride in dwords (16 + 1 TDM pad) */

// ---------------------------------------------------------------------------
// Kernel 0: analytic inverse of the affine cam2ego matrices (12 of them).
// ---------------------------------------------------------------------------
__global__ void k_inv(const float* __restrict__ c2e, float* __restrict__ e2c) {
    int bn = threadIdx.x;
    if (bn >= BN_) return;
    const float* M = c2e + bn * 16;
    float a00 = M[0], a01 = M[1], a02 = M[2],  t0 = M[3];
    float a10 = M[4], a11 = M[5], a12 = M[6],  t1 = M[7];
    float a20 = M[8], a21 = M[9], a22 = M[10], t2 = M[11];
    float c00 = a11 * a22 - a12 * a21;
    float c01 = a12 * a20 - a10 * a22;
    float c02 = a10 * a21 - a11 * a20;
    float id  = 1.0f / (a00 * c00 + a01 * c01 + a02 * c02);
    float i00 =  c00 * id;
    float i01 = (a02 * a21 - a01 * a22) * id;
    float i02 = (a01 * a12 - a02 * a11) * id;
    float i10 =  c01 * id;
    float i11 = (a00 * a22 - a02 * a20) * id;
    float i12 = (a02 * a10 - a00 * a12) * id;
    float i20 =  c02 * id;
    float i21 = (a01 * a20 - a00 * a21) * id;
    float i22 = (a00 * a11 - a01 * a10) * id;
    float* O = e2c + bn * 12;
    O[0] = i00; O[1] = i01; O[2]  = i02; O[3]  = -(i00 * t0 + i01 * t1 + i02 * t2);
    O[4] = i10; O[5] = i11; O[6]  = i12; O[7]  = -(i10 * t0 + i11 * t1 + i12 * t2);
    O[8] = i20; O[9] = i21; O[10] = i22; O[11] = -(i20 * t0 + i21 * t1 + i22 * t2);
}

// ---------------------------------------------------------------------------
// Kernel 0b: convert w_depth (123x256 f32) into f16 WMMA A-fragments, laid out
// as [otile=8][kstep=8][lane=32][elem=16] so the GEMM does one contiguous
// 32-byte load per lane per k-step. Rows 123..127 are zero padding.
// ---------------------------------------------------------------------------
__global__ __launch_bounds__(256) void k_wcvt(const float* __restrict__ w,
                                              _Float16* __restrict__ wf16) {
    const int tid = blockIdx.x * 256 + threadIdx.x;   // 0..2047, one v16h each
    if (tid >= 8 * 8 * 32) return;
    const int ot   = tid >> 8;
    const int s    = (tid >> 5) & 7;
    const int lane = tid & 31;
    const int m    = lane & 15;
    const int hi   = lane >> 4;
    const int o    = ot * 16 + m;
    const int base = s * 32 + hi * 16;
    const bool ok  = (o < OC_);
    const float* wrow = w + (size_t)(ok ? o : 0) * CIN_;
    v16h a;
#pragma unroll
    for (int i = 0; i < 8; ++i) {
        a[i]     = ok ? (_Float16)wrow[base + i]      : (_Float16)0.0f;
        a[i + 8] = ok ? (_Float16)wrow[base + 8 + i]  : (_Float16)0.0f;
    }
    *(v16h*)(wf16 + (size_t)tid * 16) = a;
}

// ---------------------------------------------------------------------------
// Kernel 1: per-pixel projection GEMM via v_wmma_f32_16x16x32_f16 + softmax.
// B tile (256 chan x 16 px f32) is fetched by the Tensor Data Mover straight
// into LDS (with 1-dword row padding to spread banks), converted to f16 once,
// then 8 waves x 8 WMMA steps consume it. A-fragments come pre-converted.
// ---------------------------------------------------------------------------
__global__ __launch_bounds__(256) void k_gemm(const float* __restrict__ img,
                                              const _Float16* __restrict__ wf16,
                                              const float* __restrict__ bias,
                                              float* __restrict__ depth_out,
                                              float* __restrict__ feat_ws) {
    __shared__ float    Braw[CIN_ * BRSTRIDE + 4]; // raw f32 tile via TDM, ~17 KB
    __shared__ _Float16 Bs[16][CIN_];              // f16 [pixel][k], 8 KB
    __shared__ float    Dt[128][16];               // [out-channel][pixel], 8 KB
    __shared__ float    red[2][16][8];             // softmax reduction scratch

    const int blk = blockIdx.x;         // 0 .. BN_*PTILES-1
    const int bn  = blk / PTILES;
    const int p0  = (blk % PTILES) * 16;
    const int t   = threadIdx.x;        // 0..255

    // ---- TDM: tensor_load_to_lds of the 16x256 f32 tile (one issue, wave 0) ----
    if (t == 0) {
        const unsigned long long ga =
            (unsigned long long)(img + (size_t)bn * CIN_ * HW_ + p0);
        const unsigned int ldsoff = (unsigned int)(size_t)&Braw[0];
        // D# group0: count=1 | lds_addr | global_addr[56:0] | type=2
        v4u g0 = { 1u, ldsoff, (unsigned int)ga,
                   (unsigned int)(ga >> 32) | (2u << 30) };
        // D# group1: data_size=4B, pad_enable, pad_interval=16DW, pad_amount=1DW;
        // tensor_dim0=16 (stride 704), tensor_dim1=256; tile = 16 x 256.
        v8i g1 = { (int)0x00D20000,        // data_size=2<<16 | pad_en | interval=3<<22
                   16 << 16,               // tensor_dim0[15:0] in bits[63:48]
                   256 << 16,              // tensor_dim1[15:0] in bits[95:80]
                   16 << 16,               // tile_dim0 in bits[127:112]
                   256,                    // tile_dim1 in bits[143:128]
                   704,                    // tensor_dim0_stride low bits
                   (int)0xC0000000,        // tensor_dim1_stride[15:0]=0xC000 (180224)
                   2 };                    // tensor_dim1_stride[47:16]
        v4i zz = { 0, 0, 0, 0 };
        v8i z8 = { 0, 0, 0, 0, 0, 0, 0, 0 };
        __builtin_amdgcn_tensor_load_to_lds(g0, g1, zz, zz, z8, 0);
    }
    __builtin_amdgcn_s_wait_tensorcnt((short)0);
    __syncthreads();

    // ---- convert Braw (f32, [chan][px], stride 17) -> Bs (f16, [px][chan]) ----
    {
        const int c0 = (t & 127) * 2;   // channel pair
        const int ph = t >> 7;          // pixel half: 0 -> px 0..7, 1 -> px 8..15
        const float* r0 = &Braw[c0 * BRSTRIDE + ph * 8];
        const float* r1 = r0 + BRSTRIDE;
#pragma unroll
        for (int j = 0; j < 8; ++j) {
            union { _Float16 h[2]; unsigned u; } pk;
            pk.h[0] = (_Float16)r0[j];
            pk.h[1] = (_Float16)r1[j];
            *(unsigned*)&Bs[ph * 8 + j][c0] = pk.u;   // packed ds_store_b32
        }
    }
    __syncthreads();

    const int wave  = t >> 5;
    const int lane  = t & 31;
    const int nn    = lane & 15;
    const int hi    = lane >> 4;
    const int obase = wave * 16;
    const _Float16* wfr = wf16 + ((size_t)wave * 8 * 32 + lane) * 16;
    __builtin_prefetch(wfr, 0, 0);      // global_prefetch_b8

    v8f acc = {};
#pragma unroll
    for (int s = 0; s < 8; ++s) {
        const v16h a = *(const v16h*)(wfr + (size_t)s * 32 * 16);
        const v16h b = *(const v16h*)&Bs[nn][s * 32 + hi * 16];
        acc = __builtin_amdgcn_wmma_f32_16x16x32_f16(
            false, a, false, b, (short)0, acc, false, false);
    }

    // ---- D tile + bias into LDS (lane L: col = L&15, rows = obase + r + 8*hi) ----
#pragma unroll
    for (int r = 0; r < 8; ++r) {
        const int o = obase + r + hi * 8;
        Dt[o][nn] = acc[r] + (o < OC_ ? bias[o] : 0.0f);
    }
    __syncthreads();

    // ---- softmax over channels 0..58 (8 threads per pixel) + feat writeback ----
    const bool act = (t < 128);
    const int  p   = t & 15;
    const int  h   = (t >> 4) & 7;
    const int  d0  = h * 8;
    const int  d1  = (d0 + 8 < DD_) ? d0 + 8 : DD_;
    if (act) {
        float pm = -3.402823e38f;
        for (int d = d0; d < d1; ++d) pm = fmaxf(pm, Dt[d][p]);
        red[0][p][h] = pm;
    }
    __syncthreads();
    if (act) {
        float mx = red[0][p][0];
#pragma unroll
        for (int i = 1; i < 8; ++i) mx = fmaxf(mx, red[0][p][i]);
        float ps = 0.0f;
        for (int d = d0; d < d1; ++d) {
            float e = expf(Dt[d][p] - mx);
            Dt[d][p] = e;
            ps += e;
        }
        red[1][p][h] = ps;
    }
    __syncthreads();
    if (act) {
        float sum = 0.0f;
#pragma unroll
        for (int i = 0; i < 8; ++i) sum += red[1][p][i];
        const float inv = 1.0f / sum;
        float* dp = depth_out + ((size_t)bn * HW_ + (p0 + p)) * DD_;
        for (int d = d0; d < d1; ++d) dp[d] = Dt[d][p] * inv;
        float* fp = feat_ws + ((size_t)bn * HW_ + (p0 + p)) * COUT_;
#pragma unroll
        for (int i = 0; i < 8; ++i) fp[h * 8 + i] = Dt[DD_ + h * 8 + i][p];
    }
}

// ---------------------------------------------------------------------------
// Kernel 2: BEV gather. One thread per (b,x,y) output column; loops over the
// 6 cameras x 7 z-slices, projects the voxel, gathers depth*feat, accumulates
// 64 channels in registers, stores once. No atomics, no zero-init needed.
// ---------------------------------------------------------------------------
__global__ __launch_bounds__(256) void k_bev(const float* __restrict__ e2c,
                                             const float* __restrict__ intr,
                                             const float* __restrict__ depth,
                                             const float* __restrict__ feat,
                                             float* __restrict__ bev) {
    const int tid = blockIdx.x * 256 + threadIdx.x;
    if (tid >= B_ * X_ * Y_) return;
    const int x = tid & (X_ - 1);
    const int y = (tid >> 7) & (Y_ - 1);
    const int b = tid >> 14;

    const float vx = (float)x * 0.8f - 51.2f;
    const float vy = (float)y * 0.8f - 51.2f;

    float acc[COUT_];
#pragma unroll
    for (int j = 0; j < COUT_; ++j) acc[j] = 0.0f;

    for (int n = 0; n < N_; ++n) {
        const int bn = b * N_ + n;
        const float* M  = e2c + bn * 12;
        const float m00 = M[0], m01 = M[1], m02 = M[2],  m03 = M[3];
        const float m10 = M[4], m11 = M[5], m12 = M[6],  m13 = M[7];
        const float m20 = M[8], m21 = M[9], m22 = M[10], m23 = M[11];
        const float* Km = intr + bn * 9;
        const float fx = Km[0], cx = Km[2], fy = Km[4], cy = Km[5];
        const float bx = m00 * vx + m01 * vy + m03;
        const float by = m10 * vx + m11 * vy + m13;
        const float bz = m20 * vx + m21 * vy + m23;

        for (int z = 0; z < Z_; ++z) {
            const float vz  = (float)z - 2.5f;
            const float cxm = bx + m02 * vz;
            const float cym = by + m12 * vz;
            const float czm = bz + m22 * vz;
            const float zs  = fmaxf(czm, 0.1f);
            const float fcx = (fx * (cxm / zs) + cx) * (1.0f / 16.0f);
            const float fcy = (fy * (cym / zs) + cy) * (1.0f / 16.0f);
            const int dbin  = (int)(czm - 1.0f);   // trunc toward zero == jnp.trunc
            const bool valid = (czm > 0.5f) &&
                               (fcx >= 0.0f) && (fcx < (float)W_) &&
                               (fcy >= 0.0f) && (fcy < (float)H_) &&
                               (dbin >= 0) && (dbin < DD_);
            if (valid) {
                const int ui  = (int)fcx;
                const int vi  = (int)fcy;
                const int pix = vi * W_ + ui;
                const float sd = depth[((size_t)bn * HW_ + pix) * DD_ + dbin];
                const float4* f4 = (const float4*)(feat + ((size_t)bn * HW_ + pix) * COUT_);
#pragma unroll
                for (int jq = 0; jq < COUT_ / 4; ++jq) {
                    const float4 f = f4[jq];
                    acc[4 * jq + 0] += sd * f.x;
                    acc[4 * jq + 1] += sd * f.y;
                    acc[4 * jq + 2] += sd * f.z;
                    acc[4 * jq + 3] += sd * f.w;
                }
            }
        }
    }
#pragma unroll
    for (int j = 0; j < COUT_; ++j)
        bev[(((size_t)b * COUT_ + j) * Y_ + y) * X_ + x] = acc[j];
}

// ---------------------------------------------------------------------------
extern "C" void kernel_launch(void* const* d_in, const int* in_sizes, int n_in,
                              void* d_out, int out_size, void* d_ws, size_t ws_size,
                              hipStream_t stream) {
    (void)in_sizes; (void)n_in; (void)out_size; (void)ws_size;
    const float* img  = (const float*)d_in[0];   // (2,6,256,16,44)
    const float* c2e  = (const float*)d_in[1];   // (2,6,4,4)
    const float* intr = (const float*)d_in[2];   // (2,6,3,3)
    const float* w    = (const float*)d_in[3];   // (123,256)
    const float* bias = (const float*)d_in[4];   // (123,)

    float* bev   = (float*)d_out;                                // (2,64,128,128)
    float* depth = bev + (size_t)B_ * COUT_ * Y_ * X_;           // (2,6,16,44,59)

    float*    feat_ws = (float*)d_ws;                            // (12,704,64) f32
    float*    e2c_ws  = feat_ws + (size_t)BN_ * HW_ * COUT_;     // (12,3,4)
    _Float16* wf16_ws = (_Float16*)(e2c_ws + 144);               // 8*8*32*16 f16, 64 KB

    k_inv <<<1, 32, 0, stream>>>(c2e, e2c_ws);
    k_wcvt<<<8, 256, 0, stream>>>(w, wf16_ws);
    k_gemm<<<BN_ * PTILES, 256, 0, stream>>>(img, wf16_ws, bias, depth, feat_ws);
    k_bev <<<(B_ * X_ * Y_) / 256, 256, 0, stream>>>(e2c_ws, intr, depth, feat_ws, bev);
}
